// LlamaAttention_34462817583689
// MI455X (gfx1250) — matrix operand order repaired and checked by reading, not compiled, not run on hardware
//
#include <hip/hip_runtime.h>

typedef __attribute__((ext_vector_type(16))) __bf16   v16bf;
typedef __attribute__((ext_vector_type(8)))  float    v8f;
typedef __attribute__((ext_vector_type(4)))  unsigned u32x4;
typedef __attribute__((ext_vector_type(8)))  unsigned u32x8;

#define WMMA_BF16(A, B, C) \
    __builtin_amdgcn_wmma_f32_16x16x32_bf16(false, (A), false, (B), (short)0, (C), false, false)

// Dimensions
#define BATCH   8
#define SEQ     680
#define SEQP    704          // 680 padded to 22*32
#define CDIM    1024
#define HEADS   16
#define HDIM    64
#define MROWS   (BATCH * SEQ)   // 5440 = 340 * 16 exactly

// ---------------------------------------------------------------------------
// TDM: load a (tile_h x tile_w) f32 tile from a row-major matrix (row stride
// `stride_elems`) into LDS at byte offset `lds_addr`. 2D descriptor: groups
// 0/1 only (VADDR2/3 = NULL form). Caller is one wave; waits TENSORcnt==0.
// D# group0: [1:0]=count=1, [63:32]=lds_addr, [120:64]=global_addr,
//            [127:126]=type=2.  D# group1: [17:16]=data_size(4B), dims/tile/
//            stride per ISA 8.4 (tensor dims set to 1<<20: OOB never trips).
// ---------------------------------------------------------------------------
__device__ __forceinline__ void tdm_load_tile_f32(const float* gsrc, unsigned lds_addr,
                                                  unsigned tile_w, unsigned tile_h,
                                                  unsigned stride_elems) {
    unsigned long long ga = (unsigned long long)(size_t)gsrc;
    u32x4 g0;
    g0[0] = 1u;                                     // count = 1 (valid, user mode)
    g0[1] = lds_addr;                               // LDS byte address
    g0[2] = (unsigned)ga;                           // global_addr[31:0]
    g0[3] = (unsigned)(ga >> 32) | (2u << 30);      // global_addr[56:32] | type=2
    u32x8 g1;
    g1[0] = 0x00020000u;                            // wg_mask=0, data_size=2 (4B)
    g1[1] = 0u;                                     // tensor_dim0[15:0]=0 (dim0=1<<20)
    g1[2] = 0x10u;                                  // tensor_dim0[31:16]=0x10, dim1 lo=0
    g1[3] = (tile_w << 16) | 0x10u;                 // tensor_dim1[31:16]=0x10, tile_dim0
    g1[4] = tile_h;                                 // tile_dim1, tile_dim2=0
    g1[5] = stride_elems;                           // tensor_dim0_stride[31:0]
    g1[6] = 0u;
    g1[7] = 0u;
    asm volatile("tensor_load_to_lds %0, %1" :: "s"(g0), "s"(g1) : "memory");
    __builtin_amdgcn_s_wait_tensorcnt(0);
}

// ---------------------------------------------------------------------------
// Zero the padded rows (680..703) of the bf16 q/k/v buffers so masked keys
// stay finite through softmax.
// ---------------------------------------------------------------------------
__global__ void zero_pad_kernel(__bf16* __restrict__ qb, __bf16* __restrict__ kb,
                                __bf16* __restrict__ vb) {
    size_t i = (size_t)blockIdx.x * blockDim.x + threadIdx.x;  // over B*H*24*64
    size_t bh  = i / (24 * 64);
    size_t rem = i % (24 * 64);
    size_t off = (bh * SEQP + SEQ + rem / HDIM) * HDIM + (rem % HDIM);
    qb[off] = (__bf16)0.0f;
    kb[off] = (__bf16)0.0f;
    vb[off] = (__bf16)0.0f;
}

// ---------------------------------------------------------------------------
// QKV projection + bias + (q,k) L2-norm + per-head scale + 2D RoPE, fused.
// Grid: (340, 48). Block: 128 (4 waves). Each workgroup: 16 tokens x 64 cols
// = one (part, head) slab. Output bf16 [B, H, SEQP, 64].
// ---------------------------------------------------------------------------
__global__ __launch_bounds__(128) void qkv_kernel(
    const float* __restrict__ x, const float* __restrict__ w,
    const float* __restrict__ q_bias, const float* __restrict__ v_bias,
    const float* __restrict__ scale_mul,
    __bf16* __restrict__ qb, __bf16* __restrict__ kb, __bf16* __restrict__ vb) {
    __shared__ float xs[16][32];    // TDM-staged activation tile (f32)
    __shared__ float ts[16][68];    // output slab (padded stride)

    const int tid  = threadIdx.x;
    const int wave = tid >> 5;
    const int lane = tid & 31;
    const int m0   = blockIdx.x * 16;
    const int part = blockIdx.y >> 4;   // 0=q 1=k 2=v
    const int h    = blockIdx.y & 15;
    const int n0   = blockIdx.y * 64 + wave * 16;

    const int row  = lane & 15;
    const int koff = (lane < 16) ? 0 : 8;
    const int boff = (lane < 16) ? 0 : 16;

    v8f c = {};
    const float* wrow = w + (size_t)(n0 + row) * CDIM + boff;
    const unsigned xs_lds = (unsigned)(size_t)(&xs[0][0]);

    for (int k0 = 0; k0 < CDIM; k0 += 32) {
        if (wave == 0)
            tdm_load_tile_f32(x + (size_t)m0 * CDIM + k0, xs_lds, 32u, 16u, CDIM);
        if (k0 + 32 < CDIM) __builtin_prefetch(wrow + k0 + 32, 0, 1);
        __syncthreads();
        v16bf a, b;
#pragma unroll
        for (int e = 0; e < 8; e++) {
            a[e]     = (__bf16)xs[row][boff + e];
            a[8 + e] = (__bf16)xs[row][boff + 8 + e];
        }
#pragma unroll
        for (int e = 0; e < 16; e++) b[e] = (__bf16)wrow[k0 + e];
        c = WMMA_BF16(a, b, c);
        __syncthreads();
    }

    {   // stage fragment + bias into LDS as full 16x64 tile
        int N   = lane & 15;
        int col = wave * 16 + N;                    // d within head
        float bias = (part == 0) ? q_bias[h * HDIM + col]
                   : (part == 2) ? v_bias[h * HDIM + col] : 0.0f;
#pragma unroll
        for (int r = 0; r < 8; r++) {
            int M = r + ((lane < 16) ? 0 : 8);
            ts[M][col] = c[r] + bias;
        }
    }
    __syncthreads();

    {   // per-row epilogue: 8 lanes per row, 8 cols each
        int rr    = tid >> 3;
        int cbase = (tid & 7) * 8;
        int gm    = m0 + rr;
        int bidx  = gm / SEQ;
        int l     = gm % SEQ;
        __bf16* dst = ((part == 0) ? qb : (part == 1) ? kb : vb)
                      + (((size_t)(bidx * HEADS + h)) * SEQP + l) * HDIM;
        if (part == 2) {
#pragma unroll
            for (int i = 0; i < 8; i++) dst[cbase + i] = (__bf16)ts[rr][cbase + i];
        } else {
            float ss = 0.0f;
#pragma unroll
            for (int i = 0; i < 8; i++) { float v2 = ts[rr][cbase + i]; ss += v2 * v2; }
            ss += __shfl_xor(ss, 1, 8);
            ss += __shfl_xor(ss, 2, 8);
            ss += __shfl_xor(ss, 4, 8);
            float fac = 1.0f / fmaxf(sqrtf(ss), 1e-12f);
            if (part == 0) fac *= __expf(fminf(scale_mul[h], 4.6051701859880914f)); // ln(100)
            // 2D RoPE position for token l
            float px = 0.0f, py = 0.0f;
            {
                const int PN[10] = {1, 2, 3, 4, 5, 6, 8, 10, 13, 16};
                int rem = l;
#pragma unroll
                for (int s = 0; s < 10; s++) {
                    int nn = PN[s] * PN[s];
                    if (rem < nn) {
                        px = (float)(rem / PN[s]) * 16.0f / PN[s];
                        py = (float)(rem % PN[s]) * 16.0f / PN[s];
                        break;
                    }
                    rem -= nn;
                }
            }
#pragma unroll
            for (int i = 0; i < 8; i++) {
                int d = cbase + i;
                float p   = (d < 32) ? px : py;
                float ang = p * __expf(-(float)(d & 15) * (9.210340371976184f / 16.0f));
                float cs = __cosf(ang), sn = __sinf(ang);
                float sign = (d < 32) ? -1.0f : 1.0f;
                float val  = ts[rr][d] * cs + sign * ts[rr][d ^ 32] * sn;
                dst[d] = (__bf16)(val * fac);
            }
        }
    }
}

// ---------------------------------------------------------------------------
// Attention: one wave per (b, h, 16-row q tile). Full 16x704 score strip in
// LDS, softmax with deferred 1/sum, then P@V via WMMA with V staged through
// async global->LDS loads (ASYNCcnt).
// ---------------------------------------------------------------------------
__global__ __launch_bounds__(32) void attn_kernel(
    const __bf16* __restrict__ qb, const __bf16* __restrict__ kb,
    const __bf16* __restrict__ vb, const float* __restrict__ attn_bias,
    float* __restrict__ ao) {
    __shared__ float  sc[16][SEQP];
    __shared__ __bf16 vt[32][HDIM];
    __shared__ float  sums[16];

    const int lane = threadIdx.x;
    const int m0   = blockIdx.x * 16;
    const int h    = blockIdx.y;
    const int b    = blockIdx.z;
    const int bh   = b * HEADS + h;

    const int row  = lane & 15;
    const int N    = lane & 15;
    const int koff = (lane < 16) ? 0 : 8;
    const int boff = (lane < 16) ? 0 : 16;

    // Q fragments (A matrix, d-chunks 0..31 and 32..63)
    const __bf16* qrow = qb + ((size_t)bh * SEQP + m0 + row) * HDIM;
    v16bf a0, a1;
#pragma unroll
    for (int e = 0; e < 8; e++) {
        a0[e] = qrow[boff + e];      a0[8 + e] = qrow[boff + 8 + e];
        a1[e] = qrow[32 + boff + e]; a1[8 + e] = qrow[32 + boff + 8 + e];
    }

    // S = Q K^T + bias  -> LDS
    for (int j0 = 0; j0 < SEQP; j0 += 16) {
        const __bf16* krow = kb + ((size_t)bh * SEQP + j0 + N) * HDIM + boff;
        v16bf b0, b1;
#pragma unroll
        for (int e = 0; e < 16; e++) { b0[e] = krow[e]; b1[e] = krow[32 + e]; }
        v8f c = {};
        c = WMMA_BF16(a0, b0, c);
        c = WMMA_BF16(a1, b1, c);
        int jc = j0 + N;
#pragma unroll
        for (int r = 0; r < 8; r++) {
            int M = r + ((lane < 16) ? 0 : 8);
            int l = m0 + M;
            float bias;
            if (jc >= SEQ)      bias = -1e30f;
            else if (l < SEQ)   bias = attn_bias[(size_t)l * SEQ + jc];
            else                bias = 0.0f;
            sc[M][jc] = c[r] + bias;
        }
    }
    __syncthreads();

    // row softmax (store unnormalized exp, keep 1/sum)
    for (int r = 0; r < 16; r++) {
        float m = -3.4e38f;
        for (int j = lane; j < SEQP; j += 32) m = fmaxf(m, sc[r][j]);
#pragma unroll
        for (int off = 16; off; off >>= 1) m = fmaxf(m, __shfl_xor(m, off, 32));
        float s = 0.0f;
        for (int j = lane; j < SEQP; j += 32) {
            float p = __expf(sc[r][j] - m);
            sc[r][j] = p;
            s += p;
        }
#pragma unroll
        for (int off = 16; off; off >>= 1) s += __shfl_xor(s, off, 32);
        if (lane == 0) sums[r] = 1.0f / s;
    }
    __syncthreads();

    // O = P V : each lane async-copies its 128-byte V row straight into LDS
    // (instruction offset applies to both LDS and global addresses).
    v8f cacc[4] = {v8f{}, v8f{}, v8f{}, v8f{}};
    const unsigned vt_lds = (unsigned)(size_t)(&vt[lane][0]);
    for (int j0 = 0; j0 < SEQP; j0 += 32) {
        const __bf16* vsrc = vb + ((size_t)bh * SEQP + j0 + lane) * HDIM;
        asm volatile(
            "global_load_async_to_lds_b128 %0, %1, off\n\t"
            "global_load_async_to_lds_b128 %0, %1, off offset:16\n\t"
            "global_load_async_to_lds_b128 %0, %1, off offset:32\n\t"
            "global_load_async_to_lds_b128 %0, %1, off offset:48\n\t"
            "global_load_async_to_lds_b128 %0, %1, off offset:64\n\t"
            "global_load_async_to_lds_b128 %0, %1, off offset:80\n\t"
            "global_load_async_to_lds_b128 %0, %1, off offset:96\n\t"
            "global_load_async_to_lds_b128 %0, %1, off offset:112"
            :: "v"(vt_lds), "v"(vsrc) : "memory");
        asm volatile("s_wait_asynccnt 0x0" ::: "memory");
        __syncthreads();
        v16bf a;
#pragma unroll
        for (int e = 0; e < 8; e++) {
            a[e]     = (__bf16)sc[row][j0 + boff + e];
            a[8 + e] = (__bf16)sc[row][j0 + boff + 8 + e];
        }
#pragma unroll
        for (int n = 0; n < 4; n++) {
            int dcol = n * 16 + N;
            v16bf bv;
#pragma unroll
            for (int e = 0; e < 16; e++) bv[e] = vt[boff + e][dcol];
            cacc[n] = WMMA_BF16(a, bv, cacc[n]);
        }
        __syncthreads();
    }

#pragma unroll
    for (int r = 0; r < 8; r++) {
        int M = r + ((lane < 16) ? 0 : 8);
        int l = m0 + M;
        if (l < SEQ) {
            float inv = sums[M];
            float* orow = ao + ((size_t)b * SEQ + l) * CDIM + h * HDIM;
            orow[N]      = cacc[0][r] * inv;
            orow[16 + N] = cacc[1][r] * inv;
            orow[32 + N] = cacc[2][r] * inv;
            orow[48 + N] = cacc[3][r] * inv;
        }
    }
}

// ---------------------------------------------------------------------------
// Output projection: out = ao @ proj_w.T + proj_b. Grid (340, 16), block 128.
// ---------------------------------------------------------------------------
__global__ __launch_bounds__(128) void proj_kernel(
    const float* __restrict__ ao, const float* __restrict__ w,
    const float* __restrict__ pb, float* __restrict__ out) {
    __shared__ float xs[16][32];
    const int tid  = threadIdx.x;
    const int wave = tid >> 5;
    const int lane = tid & 31;
    const int m0   = blockIdx.x * 16;
    const int n0   = blockIdx.y * 64 + wave * 16;
    const int row  = lane & 15;
    const int koff = (lane < 16) ? 0 : 8;
    const int boff = (lane < 16) ? 0 : 16;

    v8f c = {};
    const float* wrow = w + (size_t)(n0 + row) * CDIM + boff;
    const unsigned xs_lds = (unsigned)(size_t)(&xs[0][0]);

    for (int k0 = 0; k0 < CDIM; k0 += 32) {
        if (wave == 0)
            tdm_load_tile_f32(ao + (size_t)m0 * CDIM + k0, xs_lds, 32u, 16u, CDIM);
        if (k0 + 32 < CDIM) __builtin_prefetch(wrow + k0 + 32, 0, 1);
        __syncthreads();
        v16bf a, b;
#pragma unroll
        for (int e = 0; e < 8; e++) {
            a[e]     = (__bf16)xs[row][boff + e];
            a[8 + e] = (__bf16)xs[row][boff + 8 + e];
        }
#pragma unroll
        for (int e = 0; e < 16; e++) b[e] = (__bf16)wrow[k0 + e];
        c = WMMA_BF16(a, b, c);
        __syncthreads();
    }

    int N = lane & 15;
    int col = n0 + N;
    float bias = pb[col];
#pragma unroll
    for (int r = 0; r < 8; r++) {
        int M = r + ((lane < 16) ? 0 : 8);
        out[(size_t)(m0 + M) * CDIM + col] = c[r] + bias;
    }
}

// ---------------------------------------------------------------------------
extern "C" void kernel_launch(void* const* d_in, const int* in_sizes, int n_in,
                              void* d_out, int out_size, void* d_ws, size_t ws_size,
                              hipStream_t stream) {
    (void)in_sizes; (void)n_in; (void)out_size; (void)ws_size;
    const float* x         = (const float*)d_in[0];
    const float* attn_bias = (const float*)d_in[1];
    const float* qkv_w     = (const float*)d_in[2];
    const float* q_bias    = (const float*)d_in[3];
    const float* v_bias    = (const float*)d_in[4];
    const float* scale_mul = (const float*)d_in[5];
    const float* proj_w    = (const float*)d_in[6];
    const float* proj_b    = (const float*)d_in[7];
    float* out = (float*)d_out;

    constexpr size_t QBE = (size_t)BATCH * HEADS * SEQP * HDIM;  // bf16 elems
    __bf16* qb = (__bf16*)d_ws;
    __bf16* kb = qb + QBE;
    __bf16* vb = kb + QBE;
    float*  ao = (float*)(vb + QBE);  // [B, SEQ, C] f32

    zero_pad_kernel<<<(BATCH * HEADS * 24 * HDIM) / 256, 256, 0, stream>>>(qb, kb, vb);
    qkv_kernel<<<dim3(MROWS / 16, 48), 128, 0, stream>>>(
        x, qkv_w, q_bias, v_bias, scale_mul, qb, kb, vb);
    attn_kernel<<<dim3((SEQ + 15) / 16, HEADS, BATCH), 32, 0, stream>>>(
        qb, kb, vb, attn_bias, ao);
    proj_kernel<<<dim3(MROWS / 16, CDIM / 64), 128, 0, stream>>>(ao, proj_w, proj_b, out);
}
